// QueryRegressionLast_NEW_89051851915371
// MI455X (gfx1250) — matrix operand order repaired and checked by reading, not compiled, not run on hardware
//
#include <hip/hip_runtime.h>

// ---- problem constants (from reference) ----
#define B_      4
#define C_      512
#define T_      2304
#define TP_     2306          // T + 2 pad columns
#define KS_     3
#define NCLS_   200
#define ODIM_   2
#define LASTFC_ 3074          // 2*(512*3+1)
#define MDYN_   400           // N_CLS * OUT_DIM
#define MDYNP_  448           // padded to multiple of 64
#define FC_     512
#define OUTMAIN_ (B_*NCLS_*ODIM_*T_)   // 3686400 floats, then B_*T_ mask floats

#define TT_     64            // t-columns per block tile
#define NST_    (TT_/16)      // 4 t-subtiles per block
#define NBT_    (T_/TT_)      // 36 t-tiles
#define LDSROWS_ (TT_ + 2)    // 66
#define LDSW_    520          // 512 + 8 pad (260 dwords -> 4-bank skew per row)

typedef unsigned short u16;
typedef unsigned int   u32;
typedef __bf16 bf16x16 __attribute__((ext_vector_type(16)));
typedef float  f32x8   __attribute__((ext_vector_type(8)));
typedef u32    u32x8   __attribute__((ext_vector_type(8)));

__device__ __forceinline__ u16 f2bf(float f) {
    u32 u = __float_as_uint(f);
    u32 r = u + 0x7FFFu + ((u >> 16) & 1u);   // round-to-nearest-even
    return (u16)(r >> 16);
}

// Build zero-padded bf16 activation buffer [B][C][TP] from f32 [B][C][T]
__global__ void build_xpad_kernel(const float* __restrict__ x, u16* __restrict__ xp, int total) {
    int i = blockIdx.x * blockDim.x + threadIdx.x;
    if (i >= total) return;
    int t  = i % TP_;
    int bc = i / TP_;
    float v = (t == 0 || t == TP_ - 1) ? 0.0f : x[(size_t)bc * T_ + (t - 1)];
    xp[i] = f2bf(v);
}

// Zero pad columns (t'=0 and t'=TP-1) of a [rows][TP] bf16 buffer
__global__ void zero_pads_kernel(u16* __restrict__ yp, int rows) {
    int i = blockIdx.x * blockDim.x + threadIdx.x;
    if (i >= rows) return;
    yp[(size_t)i * TP_] = 0;
    yp[(size_t)i * TP_ + TP_ - 1] = 0;
}

// Repack conv weights (C_out, C_in, KS) f32 -> [k][m][c] bf16
__global__ void pack_w_kernel(const float* __restrict__ w, u16* __restrict__ wk) {
    int i = blockIdx.x * blockDim.x + threadIdx.x;
    if (i >= KS_ * C_ * C_) return;
    int c = i % C_;
    int rem = i / C_;
    int m = rem % C_;
    int k = rem / C_;
    wk[i] = f2bf(w[((size_t)m * C_ + c) * KS_ + k]);
}

// Repack dynamic weights from MLP output h2 [NCLS][LASTFC]:
//   wd[k][row=(n*2+o)][c] = h2[n][c*6 + o*3 + k], bd[row] = h2[n][3072+o]
__global__ void pack_dyn_kernel(const float* __restrict__ h2, u16* __restrict__ wd, float* __restrict__ bd) {
    const int WN = KS_ * MDYNP_ * C_;
    int i = blockIdx.x * blockDim.x + threadIdx.x;
    if (i < WN) {
        int c = i % C_;
        int rem = i / C_;
        int row = rem % MDYNP_;
        int k = rem / MDYNP_;
        float v = 0.0f;
        if (row < MDYN_) {
            int nn = row >> 1, o = row & 1;
            v = h2[(size_t)nn * LASTFC_ + c * 6 + o * 3 + k];
        }
        wd[i] = f2bf(v);
    } else if (i < WN + MDYNP_) {
        int row = i - WN;
        float v = 0.0f;
        if (row < MDYN_) {
            int nn = row >> 1, o = row & 1;
            v = h2[(size_t)nn * LASTFC_ + KS_ * C_ * ODIM_ + o];
        }
        bd[row] = v;
    }
}

// Simple f32 GEMM + bias (+optional relu) for the tiny query MLP
__global__ void qmm_kernel(const float* __restrict__ X, const float* __restrict__ W,
                           const float* __restrict__ bias, float* __restrict__ Y,
                           int M, int K, int N, int do_relu) {
    int i = blockIdx.x * blockDim.x + threadIdx.x;
    if (i >= M * N) return;
    int r = i / N, j = i % N;
    float s = bias[j];
    const float* xr = X + (size_t)r * K;
    for (int k = 0; k < K; ++k) s += xr[k] * W[(size_t)k * N + j];
    Y[i] = do_relu ? fmaxf(s, 0.0f) : s;
}

// In-place LayerNorm (+relu) over FC_ features per row; one block per row
__global__ __launch_bounds__(256)
void ln_relu_kernel(float* __restrict__ X, const float* __restrict__ g, const float* __restrict__ bta) {
    __shared__ float ssum[256];
    __shared__ float ssq[256];
    int r = blockIdx.x;
    float* xr = X + (size_t)r * FC_;
    float lsum = 0.f, lsq = 0.f;
    for (int j = threadIdx.x; j < FC_; j += 256) { float v = xr[j]; lsum += v; lsq += v * v; }
    ssum[threadIdx.x] = lsum; ssq[threadIdx.x] = lsq;
    __syncthreads();
    for (int s = 128; s > 0; s >>= 1) {
        if (threadIdx.x < (unsigned)s) { ssum[threadIdx.x] += ssum[threadIdx.x + s]; ssq[threadIdx.x] += ssq[threadIdx.x + s]; }
        __syncthreads();
    }
    float mu  = ssum[0] * (1.0f / FC_);
    float var = ssq[0] * (1.0f / FC_) - mu * mu;
    float inv = rsqrtf(var + 1e-5f);
    for (int j = threadIdx.x; j < FC_; j += 256) {
        float v = (xr[j] - mu) * inv * g[j] + bta[j];
        xr[j] = fmaxf(v, 0.0f);
    }
}

// Append mask tuple output as floats
__global__ void mask_out_kernel(const unsigned char* __restrict__ mask, float* __restrict__ out) {
    int i = blockIdx.x * blockDim.x + threadIdx.x;
    if (i >= B_ * T_) return;
    out[OUTMAIN_ + i] = mask[i] ? 1.0f : 0.0f;
}

// Conv-as-GEMM via WMMA bf16 with LDS-staged, transposed activation strip.
// Block tile: full Mpad (<=512) x 64 t-columns. 8 waves; wave w = 64-row M quad.
// Y[b][m][t] = act( (sum_{tap,c} Wk[tap][m][c]*Xpad[b][c][t+tap]) + bias[m] )
// If Ypad != null: store relu(v*mask) as bf16 into padded buffer (conv layers).
// Else:            store relu(scale*v*mask) as f32 into Yf32 [b][Mvalid][T] (final einsum).
__global__ __launch_bounds__(256)
void conv_wmma_kernel(const u16* __restrict__ Xpad, const u16* __restrict__ Wk,
                      const float* __restrict__ bias, const unsigned char* __restrict__ mask,
                      u16* __restrict__ Ypad, float* __restrict__ Yf32,
                      const float* __restrict__ scale_ptr,
                      int Mpad, int Mvalid) {
    __shared__ u16 xt[LDSROWS_ * LDSW_];   // xt[tloc][c], row stride 520 u16

    const int b  = blockIdx.x / NBT_;
    const int tt = blockIdx.x % NBT_;
    const int t0 = tt * TT_;
    const int lane = threadIdx.x & 31;
    const int wave = threadIdx.x >> 5;

    // ---- stage Xpad[b][0..511][t0 .. t0+TT+1] into LDS transposed ----
    {
        const int c0 = threadIdx.x * 2;        // rows c0, c0+1
        const u32* g0 = (const u32*)(Xpad + (size_t)(b * C_ + c0) * TP_ + t0);
        const u32* g1 = (const u32*)(Xpad + (size_t)(b * C_ + c0 + 1) * TP_ + t0);
#pragma unroll
        for (int i = 0; i < LDSROWS_ / 2; ++i) {
            u32 a0 = g0[i];                    // {t=2i, t=2i+1} of row c0
            u32 a1 = g1[i];                    // {t=2i, t=2i+1} of row c0+1
            *(u32*)&xt[(2 * i)     * LDSW_ + c0] = (a0 & 0xFFFFu) | (a1 << 16);
            *(u32*)&xt[(2 * i + 1) * LDSW_ + c0] = (a0 >> 16)     | (a1 & 0xFFFF0000u);
        }
    }
    __syncthreads();

    const int mquads = Mpad >> 6;
    if (wave >= mquads) return;                // uniform per wave, after the barrier
    const int m0 = wave << 6;
    const int half = lane >> 4;                // 0/1
    const int n    = lane & 15;

    f32x8 acc[4][NST_] = {};

    for (int tap = 0; tap < KS_; ++tap) {
        const u16* wbase = Wk + (size_t)tap * Mpad * C_;
        for (int cb = 0; cb < C_; cb += 32) {
            // B fragments (32x16) for the t-subtiles: lane n holds K = 16h..16h+15,
            // i.e. 16 contiguous c values at xt[s*16+n+tap][cb+16h .. +15]
            bf16x16 bfr[NST_];
#pragma unroll
            for (int s = 0; s < NST_; ++s) {
                const uint4* p = (const uint4*)&xt[(size_t)(s * 16 + n + tap) * LDSW_ + cb + 16 * half];
                uint4 lo = p[0], hi = p[1];
                u32x8 bu;
                bu[0] = lo.x; bu[1] = lo.y; bu[2] = lo.z; bu[3] = lo.w;
                bu[4] = hi.x; bu[5] = hi.y; bu[6] = hi.z; bu[7] = hi.w;
                bfr[s] = __builtin_bit_cast(bf16x16, bu);
            }
            // A fragments (16x32): v0-7 hold K=16h..16h+15 (contiguous, matching B)
#pragma unroll
            for (int q = 0; q < 4; ++q) {
                const u16* wm = wbase + (size_t)(m0 + q * 16 + n) * C_ + cb + half * 16;
                const uint4* p = (const uint4*)wm;
                uint4 lo = p[0];
                uint4 hi = p[1];
                u32x8 au;
                au[0] = lo.x; au[1] = lo.y; au[2] = lo.z; au[3] = lo.w;
                au[4] = hi.x; au[5] = hi.y; au[6] = hi.z; au[7] = hi.w;
                bf16x16 af = __builtin_bit_cast(bf16x16, au);
#pragma unroll
                for (int s = 0; s < NST_; ++s) {
                    acc[q][s] = __builtin_amdgcn_wmma_f32_16x16x32_bf16(
                        false, af, false, bfr[s], (short)0, acc[q][s], false, false);
                }
            }
        }
    }

    const float sc = scale_ptr ? scale_ptr[0] : 1.0f;
#pragma unroll
    for (int s = 0; s < NST_; ++s) {
        const int tcol = t0 + s * 16 + n;
        const float mv = mask[(size_t)b * T_ + tcol] ? 1.0f : 0.0f;
#pragma unroll
        for (int q = 0; q < 4; ++q) {
#pragma unroll
            for (int j = 0; j < 8; ++j) {
                int m = m0 + q * 16 + half * 8 + j;   // C/D: lanes 0-15 rows 0-7, 16-31 rows 8-15
                float v = acc[q][s][j] + bias[m];
                if (Ypad) {
                    Ypad[((size_t)b * Mpad + m) * TP_ + 1 + tcol] = f2bf(fmaxf(v * mv, 0.0f));
                } else if (m < Mvalid) {
                    Yf32[((size_t)b * Mvalid + m) * T_ + tcol] = fmaxf(sc * v * mv, 0.0f);
                }
            }
        }
    }
}

extern "C" void kernel_launch(void* const* d_in, const int* in_sizes, int n_in,
                              void* d_out, int out_size, void* d_ws, size_t ws_size,
                              hipStream_t stream) {
    (void)in_sizes; (void)n_in; (void)out_size; (void)ws_size;
    const float* fpn_feat         = (const float*)d_in[0];
    const unsigned char* fpn_mask = (const unsigned char*)d_in[1];
    const float* query = (const float*)d_in[2];
    const float* hw0   = (const float*)d_in[3];
    const float* hb0   = (const float*)d_in[4];
    const float* hw1   = (const float*)d_in[5];
    const float* hb1   = (const float*)d_in[6];
    const float* fw0   = (const float*)d_in[7];
    const float* fb0   = (const float*)d_in[8];
    const float* lg0   = (const float*)d_in[9];
    const float* lb0   = (const float*)d_in[10];
    const float* fw1   = (const float*)d_in[11];
    const float* fb1   = (const float*)d_in[12];
    const float* lg1   = (const float*)d_in[13];
    const float* lb1   = (const float*)d_in[14];
    const float* fw2   = (const float*)d_in[15];
    const float* fb2   = (const float*)d_in[16];
    const float* scale = (const float*)d_in[17];
    float* out = (float*)d_out;

    char* ws = (char*)d_ws;
    size_t off = 0;
    auto take = [&](size_t bytes) -> char* {
        char* p = ws + off;
        off = (off + bytes + 255) & ~(size_t)255;
        return p;
    };
    u16*   xpad = (u16*)take((size_t)B_ * C_ * TP_ * 2);
    u16*   y1   = (u16*)take((size_t)B_ * C_ * TP_ * 2);
    u16*   y2   = (u16*)take((size_t)B_ * C_ * TP_ * 2);
    u16*   w0k  = (u16*)take((size_t)KS_ * C_ * C_ * 2);
    u16*   w1k  = (u16*)take((size_t)KS_ * C_ * C_ * 2);
    u16*   wd   = (u16*)take((size_t)KS_ * MDYNP_ * C_ * 2);
    float* bd   = (float*)take((size_t)MDYNP_ * 4);
    float* h0   = (float*)take((size_t)NCLS_ * FC_ * 4);
    float* h1   = (float*)take((size_t)NCLS_ * FC_ * 4);
    float* h2   = (float*)take((size_t)NCLS_ * LASTFC_ * 4);

    // Stage inputs
    { int n = B_ * C_ * TP_; build_xpad_kernel<<<(n + 255) / 256, 256, 0, stream>>>(fpn_feat, xpad, n); }
    { int n = KS_ * C_ * C_;
      pack_w_kernel<<<(n + 255) / 256, 256, 0, stream>>>(hw0, w0k);
      pack_w_kernel<<<(n + 255) / 256, 256, 0, stream>>>(hw1, w1k); }
    { int n = B_ * C_;
      zero_pads_kernel<<<(n + 255) / 256, 256, 0, stream>>>(y1, n);
      zero_pads_kernel<<<(n + 255) / 256, 256, 0, stream>>>(y2, n); }

    // Two conv layers via WMMA (block = full 512 M x 64 T)
    { int blocks = B_ * NBT_;
      conv_wmma_kernel<<<blocks, 256, 0, stream>>>(xpad, w0k, hb0, fpn_mask, y1, nullptr, nullptr, C_, C_);
      conv_wmma_kernel<<<blocks, 256, 0, stream>>>(y1,   w1k, hb1, fpn_mask, y2, nullptr, nullptr, C_, C_); }

    // Query MLP (tiny, exact f32)
    { int n = NCLS_ * FC_; qmm_kernel<<<(n + 255) / 256, 256, 0, stream>>>(query, fw0, fb0, h0, NCLS_, FC_, FC_, 0); }
    ln_relu_kernel<<<NCLS_, 256, 0, stream>>>(h0, lg0, lb0);
    { int n = NCLS_ * FC_; qmm_kernel<<<(n + 255) / 256, 256, 0, stream>>>(h0, fw1, fb1, h1, NCLS_, FC_, FC_, 0); }
    ln_relu_kernel<<<NCLS_, 256, 0, stream>>>(h1, lg1, lb1);
    { int n = NCLS_ * LASTFC_; qmm_kernel<<<(n + 255) / 256, 256, 0, stream>>>(h1, fw2, fb2, h2, NCLS_, FC_, LASTFC_, 1); }
    { int n = KS_ * MDYNP_ * C_ + MDYNP_; pack_dyn_kernel<<<(n + 255) / 256, 256, 0, stream>>>(h2, wd, bd); }

    // Dynamic conv (einsum) via WMMA -> final output (wave 7 idle: Mpad=448)
    { int blocks = B_ * NBT_;
      conv_wmma_kernel<<<blocks, 256, 0, stream>>>(y2, wd, bd, fpn_mask, nullptr, out, scale, MDYNP_, MDYN_); }

    // Mask tuple element
    { int n = B_ * T_; mask_out_kernel<<<(n + 255) / 256, 256, 0, stream>>>(fpn_mask, out); }
}
